// MoELayer_Loss_44736379355513
// MI455X (gfx1250) — compile-verified, matrix operand order checked by reading, not run on hardware
//
#include <hip/hip_runtime.h>

// ---------------------------------------------------------------------------
// MoE (top-2 of 8 experts), DIM=1024, HID=4096, T=4096 tokens.
// bf16 WMMA (v_wmma_f32_16x16x32_bf16) grouped GEMM, fp32 accumulate.
// Tiles: 64(M) x 128(N) x 64(K). 8 waves/block, 32x32 patch per wave ->
// 8 WMMAs per staged K-tile per wave. Register-prefetched staging with
// moving-pointer addressing (immediate-offset global loads).
// Workspace (~67.6 MB):
//   [0,32) int cnt[8] | [32,64) f_i[8] | [64,96) P_i[8]
//   [128)    int   elist[8*4096]  (pair ids = token*2 + k)
//   [+128KB) float evals[8*4096]
//   [512KB)  bf16  H[8192*4096]
// ---------------------------------------------------------------------------

#define DIM   1024
#define HID   4096
#define NEXP  8
#define TOPK  2
#define T_TOK 4096

#define TM 64
#define TN 128
#define TK 64
#define LSTR 72   // LDS row stride in bf16 elems: 144B (16B-multiple)

typedef __bf16 bf16;
typedef __attribute__((ext_vector_type(4)))  __bf16 v4bf;
typedef __attribute__((ext_vector_type(16))) __bf16 v16bf;
typedef __attribute__((ext_vector_type(8)))  float  v8f;

__device__ __forceinline__ void atomAddF(float* p, float v) {
  unsafeAtomicAdd(p, v);   // global_atomic_add_f32
}

// ------------------------------- gating ------------------------------------
__global__ __launch_bounds__(256)
void gate_kernel(const float* __restrict__ u, const float* __restrict__ cent,
                 int* __restrict__ cnt, float* __restrict__ f_i,
                 float* __restrict__ P_i, int* __restrict__ elist,
                 float* __restrict__ evals) {
  __shared__ float cs[NEXP * DIM];            // 32 KB
  const int tid = threadIdx.x;
  for (int i = tid; i < NEXP * DIM; i += 256) cs[i] = cent[i];
  __syncthreads();

  const int t = blockIdx.x * 256 + tid;
  const float* ur = u + (size_t)t * DIM;

  float acc[NEXP];
#pragma unroll
  for (int e = 0; e < NEXP; ++e) acc[e] = 0.f;

  for (int d = 0; d < DIM; d += 4) {
    const float4 uv = *(const float4*)(ur + d);
#pragma unroll
    for (int e = 0; e < NEXP; ++e) {
      const float4 cv = *(const float4*)(cs + e * DIM + d);
      acc[e] += uv.x * cv.x + uv.y * cv.y + uv.z * cv.z + uv.w * cv.w;
    }
  }

  float mx = acc[0];
#pragma unroll
  for (int e = 1; e < NEXP; ++e) mx = fmaxf(mx, acc[e]);
  float ex[NEXP], den = 0.f;
#pragma unroll
  for (int e = 0; e < NEXP; ++e) { ex[e] = __expf(acc[e] - mx); den += ex[e]; }
  const float rden = 1.f / den;

  int i0 = 0;
#pragma unroll
  for (int e = 1; e < NEXP; ++e) if (acc[e] > acc[i0]) i0 = e;
  int i1 = (i0 == 0) ? 1 : 0;
#pragma unroll
  for (int e = 0; e < NEXP; ++e) if (e != i0 && acc[e] > acc[i1]) i1 = e;

  const float v0 = ex[i0] * rden, v1 = ex[i1] * rden;

  int s0 = atomicAdd(&cnt[i0], 1);
  elist[i0 * T_TOK + s0] = t * 2 + 0;
  evals[i0 * T_TOK + s0] = v0;
  atomAddF(&f_i[i0], 1.f);  atomAddF(&P_i[i0], v0);

  int s1 = atomicAdd(&cnt[i1], 1);
  elist[i1 * T_TOK + s1] = t * 2 + 1;
  evals[i1 * T_TOK + s1] = v1;
  atomAddF(&f_i[i1], 1.f);  atomAddF(&P_i[i1], v1);
}

// ------------------------------ aux loss -----------------------------------
__global__ void loss_kernel(const float* __restrict__ f_i,
                            const float* __restrict__ P_i,
                            float* __restrict__ tail) {
  const int l = threadIdx.x;                  // 32 lanes (wave32)
  const float f = (l < NEXP) ? f_i[l] : 0.f;
  const float p = (l < NEXP) ? P_i[l] : 0.f;
  float contrib = f * ((float)NEXP / (float)(TOPK * T_TOK)) * (p / (float)T_TOK);
  float mx = f;
#pragma unroll
  for (int off = 16; off > 0; off >>= 1) {
    mx = fmaxf(mx, __shfl_xor(mx, off));
    contrib += __shfl_xor(contrib, off);
  }
  if (l == 0) {
    const float perfect = (float)(TOPK * T_TOK) / NEXP;
    tail[0] = (mx - perfect) / perfect;
    tail[1] = 0.01f * contrib;
  }
}

// ----------------- shared WMMA micro-kernel (per staged tile) --------------
__device__ __forceinline__ void wmma_tile(const bf16* __restrict__ Xs,
                                          const bf16* __restrict__ Wt,
                                          int wm, int wn, int lr, int klo,
                                          v8f& c00, v8f& c01, v8f& c10, v8f& c11) {
#pragma unroll
  for (int kk = 0; kk < TK; kk += 32) {
    union { v16bf v; uint4 q[2]; } a0, a1, b0, b1f;
    const bf16* a0p = &Xs[(wm * 32 + lr) * LSTR + kk];
    const bf16* a1p = &Xs[(wm * 32 + 16 + lr) * LSTR + kk];
    a0.q[0] = *(const uint4*)(a0p + klo);  a0.q[1] = *(const uint4*)(a0p + 16 + klo);
    a1.q[0] = *(const uint4*)(a1p + klo);  a1.q[1] = *(const uint4*)(a1p + 16 + klo);
    const bf16* b0p = &Wt[(wn * 32 + lr) * LSTR + kk];
    const bf16* b1p = &Wt[(wn * 32 + 16 + lr) * LSTR + kk];
    b0.q[0]  = *(const uint4*)(b0p + klo); b0.q[1]  = *(const uint4*)(b0p + 16 + klo);
    b1f.q[0] = *(const uint4*)(b1p + klo); b1f.q[1] = *(const uint4*)(b1p + 16 + klo);

    c00 = __builtin_amdgcn_wmma_f32_16x16x32_bf16(false, a0.v, false, b0.v,  (short)0, c00, false, false);
    c01 = __builtin_amdgcn_wmma_f32_16x16x32_bf16(false, a0.v, false, b1f.v, (short)0, c01, false, false);
    c10 = __builtin_amdgcn_wmma_f32_16x16x32_bf16(false, a1.v, false, b0.v,  (short)0, c10, false, false);
    c11 = __builtin_amdgcn_wmma_f32_16x16x32_bf16(false, a1.v, false, b1f.v, (short)0, c11, false, false);
  }
}

// --------------------------- FFN pass 1: H = relu(X W1 + b1) ---------------
__global__ __launch_bounds__(256)
void ffn1_kernel(const float* __restrict__ u, const float* __restrict__ W1,
                 const float* __restrict__ b1, const int* __restrict__ cnt,
                 const int* __restrict__ elist, bf16* __restrict__ H) {
  const int e  = blockIdx.z;
  const int n0 = blockIdx.x * TN;             // over HID
  const int m0 = blockIdx.y * TM;
  const int count = cnt[e];
  if (m0 >= count) return;

  __shared__ __align__(16) bf16 Xs[TM * LSTR];
  __shared__ __align__(16) bf16 Wt[TN * LSTR];   // [n][k], transposed
  __shared__ int pair_s[TM];

  const int tid = threadIdx.x;
  if (tid < TM) {
    const int idx = m0 + tid;
    pair_s[tid] = (idx < count) ? elist[e * T_TOK + idx] : -1;
  }
  __syncthreads();

  // X: 4 float4-slots; slot = tid + s*256 (0..1023): r = slot>>4, c4 = (slot&15)*4
  const float* x_gp[4];  int x_ld[4];  bool x_ok[4];
#pragma unroll
  for (int s = 0; s < 4; ++s) {
    const int slot = tid + s * 256;
    const int r = slot >> 4, c4 = (slot & 15) * 4;
    const int pr = pair_s[r];
    x_ok[s] = pr >= 0;
    x_gp[s] = u + (size_t)(x_ok[s] ? (pr >> 1) : 0) * DIM + c4;
    x_ld[s] = r * LSTR + c4;
  }
  // W: 8 slots; slot = tid + s*256 (0..2047): cn = slot&127, rk = (slot>>7)*4
  const float* w_gp[8];  int w_ld[8];
#pragma unroll
  for (int s = 0; s < 8; ++s) {
    const int slot = tid + s * 256;
    const int cn = slot & 127, rk = (slot >> 7) * 4;
    w_gp[s] = W1 + (size_t)e * DIM * HID + (size_t)rk * HID + n0 + cn;
    w_ld[s] = cn * LSTR + rk;
  }

  const int wave = tid >> 5, lane = tid & 31;
  const int wm = wave >> 2, wn = wave & 3;    // 2x4 wave grid, 32x32 patch
  const int lr = lane & 15, hi = lane >> 4;
  const int klo = hi * 8;

  v8f c00 = {}, c01 = {}, c10 = {}, c11 = {};
  float4 xr[4];
  float  wr[8][4];

  // prefetch tile 0 (immediate-offset loads from moving pointers)
#pragma unroll
  for (int s = 0; s < 4; ++s) {
    float4 v = *(const float4*)(x_gp[s]);
    if (!x_ok[s]) v = make_float4(0.f, 0.f, 0.f, 0.f);
    xr[s] = v;  x_gp[s] += TK;
  }
#pragma unroll
  for (int s = 0; s < 8; ++s) {
#pragma unroll
    for (int i = 0; i < 4; ++i) wr[s][i] = w_gp[s][(size_t)i * HID];
    w_gp[s] += (size_t)TK * HID;
  }

  for (int kb = 0; kb < DIM; kb += TK) {      // 16 iterations
    __syncthreads();
#pragma unroll
    for (int s = 0; s < 4; ++s) {
      v4bf p = { (bf16)xr[s].x, (bf16)xr[s].y, (bf16)xr[s].z, (bf16)xr[s].w };
      *(v4bf*)&Xs[x_ld[s]] = p;
    }
#pragma unroll
    for (int s = 0; s < 8; ++s) {
      v4bf p = { (bf16)wr[s][0], (bf16)wr[s][1], (bf16)wr[s][2], (bf16)wr[s][3] };
      *(v4bf*)&Wt[w_ld[s]] = p;
    }
    __syncthreads();

    if (kb + TK < DIM) {                      // prefetch next tile
#pragma unroll
      for (int s = 0; s < 4; ++s) {
        float4 v = *(const float4*)(x_gp[s]);
        if (!x_ok[s]) v = make_float4(0.f, 0.f, 0.f, 0.f);
        xr[s] = v;  x_gp[s] += TK;
      }
#pragma unroll
      for (int s = 0; s < 8; ++s) {
#pragma unroll
        for (int i = 0; i < 4; ++i) wr[s][i] = w_gp[s][(size_t)i * HID];
        w_gp[s] += (size_t)TK * HID;
      }
    }

    wmma_tile(Xs, Wt, wm, wn, lr, klo, c00, c01, c10, c11);
  }

  // epilogue: C layout — lane l, vgpr v -> row v + 8*(l>=16), col l%16
  const v8f* cc[4] = { &c00, &c01, &c10, &c11 };
#pragma unroll
  for (int ai = 0; ai < 2; ++ai) {
#pragma unroll
    for (int bi = 0; bi < 2; ++bi) {
      const v8f& c = *cc[ai * 2 + bi];
      const int j = n0 + wn * 32 + bi * 16 + lr;
      const float bias = b1[e * HID + j];
#pragma unroll
      for (int v = 0; v < 8; ++v) {
        const int mrow = wm * 32 + ai * 16 + v + 8 * hi;
        const int pr = pair_s[mrow];
        if (pr < 0) continue;
        H[(size_t)pr * HID + j] = (bf16)fmaxf(c[v] + bias, 0.f);
      }
    }
  }
}

// ------------------- FFN pass 2: out += gate * (H W2 + b2) -----------------
__global__ __launch_bounds__(256)
void ffn2_kernel(const bf16* __restrict__ H, const float* __restrict__ W2,
                 const float* __restrict__ b2, const int* __restrict__ cnt,
                 const int* __restrict__ elist, const float* __restrict__ evals,
                 float* __restrict__ out) {
  const int e  = blockIdx.z;
  const int n0 = blockIdx.x * TN;             // over DIM
  const int m0 = blockIdx.y * TM;
  const int count = cnt[e];
  if (m0 >= count) return;

  __shared__ __align__(16) bf16 Xs[TM * LSTR];
  __shared__ __align__(16) bf16 Wt[TN * LSTR];
  __shared__ int   pair_s[TM];
  __shared__ float val_s[TM];

  const int tid = threadIdx.x;
  if (tid < TM) {
    const int idx = m0 + tid;
    const bool ok = idx < count;
    pair_s[tid] = ok ? elist[e * T_TOK + idx] : -1;
    val_s[tid]  = ok ? evals[e * T_TOK + idx] : 0.f;
  }
  __syncthreads();

  // X (bf16 H rows): 2 uint4-slots; slot = tid + s*256 (0..511): r = slot>>3, k8 = (slot&7)*8
  const bf16* x_gp[2];  int x_ld[2];  bool x_ok[2];
#pragma unroll
  for (int s = 0; s < 2; ++s) {
    const int slot = tid + s * 256;
    const int r = slot >> 3, k8 = (slot & 7) * 8;
    const int pr = pair_s[r];
    x_ok[s] = pr >= 0;
    x_gp[s] = H + (size_t)(x_ok[s] ? pr : 0) * HID + k8;
    x_ld[s] = r * LSTR + k8;
  }
  const float* w_gp[8];  int w_ld[8];
#pragma unroll
  for (int s = 0; s < 8; ++s) {
    const int slot = tid + s * 256;
    const int cn = slot & 127, rk = (slot >> 7) * 4;
    w_gp[s] = W2 + (size_t)e * HID * DIM + (size_t)rk * DIM + n0 + cn;
    w_ld[s] = cn * LSTR + rk;
  }

  const int wave = tid >> 5, lane = tid & 31;
  const int wm = wave >> 2, wn = wave & 3;
  const int lr = lane & 15, hi = lane >> 4;
  const int klo = hi * 8;

  v8f c00 = {}, c01 = {}, c10 = {}, c11 = {};
  uint4 xr[2];
  float wr[8][4];

#pragma unroll
  for (int s = 0; s < 2; ++s) {
    uint4 v = *(const uint4*)(x_gp[s]);
    if (!x_ok[s]) v = make_uint4(0u, 0u, 0u, 0u);
    xr[s] = v;  x_gp[s] += TK;
  }
#pragma unroll
  for (int s = 0; s < 8; ++s) {
#pragma unroll
    for (int i = 0; i < 4; ++i) wr[s][i] = w_gp[s][(size_t)i * DIM];
    w_gp[s] += (size_t)TK * DIM;
  }

  for (int kb = 0; kb < HID; kb += TK) {      // 64 iterations
    __syncthreads();
#pragma unroll
    for (int s = 0; s < 2; ++s) *(uint4*)&Xs[x_ld[s]] = xr[s];
#pragma unroll
    for (int s = 0; s < 8; ++s) {
      v4bf p = { (bf16)wr[s][0], (bf16)wr[s][1], (bf16)wr[s][2], (bf16)wr[s][3] };
      *(v4bf*)&Wt[w_ld[s]] = p;
    }
    __syncthreads();

    if (kb + TK < HID) {
#pragma unroll
      for (int s = 0; s < 2; ++s) {
        uint4 v = *(const uint4*)(x_gp[s]);
        if (!x_ok[s]) v = make_uint4(0u, 0u, 0u, 0u);
        xr[s] = v;  x_gp[s] += TK;
      }
#pragma unroll
      for (int s = 0; s < 8; ++s) {
#pragma unroll
        for (int i = 0; i < 4; ++i) wr[s][i] = w_gp[s][(size_t)i * DIM];
        w_gp[s] += (size_t)TK * DIM;
      }
    }

    wmma_tile(Xs, Wt, wm, wn, lr, klo, c00, c01, c10, c11);
  }

  const v8f* cc[4] = { &c00, &c01, &c10, &c11 };
#pragma unroll
  for (int ai = 0; ai < 2; ++ai) {
#pragma unroll
    for (int bi = 0; bi < 2; ++bi) {
      const v8f& c = *cc[ai * 2 + bi];
      const int d = n0 + wn * 32 + bi * 16 + lr;
      const float bias = b2[e * DIM + d];
#pragma unroll
      for (int v = 0; v < 8; ++v) {
        const int mrow = wm * 32 + ai * 16 + v + 8 * hi;
        const int pr = pair_s[mrow];
        if (pr < 0) continue;
        atomAddF(&out[(size_t)(pr >> 1) * DIM + d], val_s[mrow] * (c[v] + bias));
      }
    }
  }
}

// ------------------------------- launch ------------------------------------
extern "C" void kernel_launch(void* const* d_in, const int* in_sizes, int n_in,
                              void* d_out, int out_size, void* d_ws, size_t ws_size,
                              hipStream_t stream) {
  const float* u    = (const float*)d_in[0];
  const float* cent = (const float*)d_in[1];
  const float* W1   = (const float*)d_in[2];
  const float* b1   = (const float*)d_in[3];
  const float* W2   = (const float*)d_in[4];
  const float* b2   = (const float*)d_in[5];
  float* out = (float*)d_out;

  char* ws = (char*)d_ws;
  int*   cnt   = (int*)ws;
  float* f_i   = (float*)(ws + 32);
  float* P_i   = (float*)(ws + 64);
  int*   elist = (int*)(ws + 128);
  float* evals = (float*)(ws + 128 + NEXP * T_TOK * sizeof(int));
  bf16*  H     = (bf16*)(ws + (size_t)(1 << 19));   // 8192 x 4096 bf16

  hipMemsetAsync(ws, 0, 128, stream);
  hipMemsetAsync(out, 0, (size_t)out_size * sizeof(float), stream);

  gate_kernel<<<T_TOK / 256, 256, 0, stream>>>(u, cent, cnt, f_i, P_i, elist, evals);
  loss_kernel<<<1, 32, 0, stream>>>(f_i, P_i, out + (out_size - 2));
  ffn1_kernel<<<dim3(HID / TN, T_TOK / TM, NEXP), 256, 0, stream>>>(u, W1, b1, cnt, elist, H);
  ffn2_kernel<<<dim3(DIM / TN, T_TOK / TM, NEXP), 256, 0, stream>>>(H, W2, b2, cnt, elist, evals, out);
}
